// MambaAudioClassifier_30648886624686
// MI455X (gfx1250) — matrix operand order repaired and checked by reading, not compiled, not run on hardware
//
#include <hip/hip_runtime.h>
#include <hip/hip_bf16.h>

#define D_MODEL 256
#define PATCH   160
#define SEQ     1000
#define BATCH   32
#define NPOS    (BATCH*SEQ)   /* 32000 rows */
#define G3D     768
#define NLAYERS 6

typedef __attribute__((ext_vector_type(16))) __bf16 v16bf;
typedef __attribute__((ext_vector_type(8)))  __bf16 v8bf;
typedef __attribute__((ext_vector_type(8)))  float  v8f;
typedef __attribute__((ext_vector_type(4))) unsigned int u32x4;
typedef __attribute__((ext_vector_type(8))) int          i32x8;
typedef __attribute__((ext_vector_type(4))) int          i32x4;

__device__ __forceinline__ float gelu_exact(float x) {
    return 0.5f * x * (1.0f + erff(x * 0.70710678118654752440f));
}
__device__ __forceinline__ float sigmoid_f(float x) {
    return 1.0f / (1.0f + expf(-x));
}

// LDS byte offset of a generic pointer to __shared__ (flat shared addr low bits)
__device__ __forceinline__ uint32_t lds_off_u32(const void* p) {
    return (uint32_t)(uintptr_t)p;
}

// ---------------------------------------------------------------------------
// TDM: issue one tensor_load_to_lds for a (up to 3D) bf16 tile.
// Descriptor bit-packing per CDNA5 ISA.
// This toolchain exposes the 6-arg builtin: (u32x4, i32x8, i32x4, i32x4, i32x8, cpol)
// ---------------------------------------------------------------------------
__device__ __forceinline__ void tdm_load_bf16(
        uint32_t lds_off, uint64_t gaddr,
        uint32_t tile_d0, uint32_t tile_d1, uint32_t tile_d2,
        uint32_t ten_d0, uint32_t ten_d1, uint32_t ten_d2,
        uint64_t stride0, uint64_t stride1) {
    u32x4 g0; i32x8 g1; i32x4 g2, g3; i32x8 g4;
    g0[0] = 1u;                                    // count=1, user descriptor
    g0[1] = lds_off;                               // lds_addr (bytes)
    g0[2] = (unsigned)(gaddr & 0xFFFFFFFFu);       // global_addr[31:0]
    g0[3] = (unsigned)((gaddr >> 32) & 0x01FFFFFFu) | 0x80000000u;  // addr[56:32] | type=2
    g1[0] = (int)(1u << 16);                       // workgroup_mask=0, data_size=1 (2B)
    g1[1] = (int)((ten_d0 & 0xFFFFu) << 16);
    g1[2] = (int)(((ten_d0 >> 16) & 0xFFFFu) | ((ten_d1 & 0xFFFFu) << 16));
    g1[3] = (int)(((ten_d1 >> 16) & 0xFFFFu) | ((tile_d0 & 0xFFFFu) << 16));
    g1[4] = (int)((tile_d1 & 0xFFFFu) | ((tile_d2 & 0xFFFFu) << 16));
    g1[5] = (int)(unsigned)(stride0 & 0xFFFFFFFFu);
    g1[6] = (int)(((stride0 >> 32) & 0xFFFFu) | ((stride1 & 0xFFFFu) << 16));
    g1[7] = (int)(unsigned)((stride1 >> 16) & 0xFFFFFFFFu);
    g2[0] = (int)ten_d2; g2[1] = 0; g2[2] = 0; g2[3] = 0;
    g3[0] = 0; g3[1] = 0; g3[2] = 0; g3[3] = 0;
    g4[0] = 0; g4[1] = 0; g4[2] = 0; g4[3] = 0;
    g4[4] = 0; g4[5] = 0; g4[6] = 0; g4[7] = 0;
    __builtin_amdgcn_tensor_load_to_lds(g0, g1, g2, g3, g4, 0);
}

// ---------------------------------------------------------------------------
// Weight conversion kernels (run once per launch; weights stay L2-resident)
// ---------------------------------------------------------------------------
__global__ void cvt_convw(const float* __restrict__ w, __bf16* __restrict__ o) {
    int i = blockIdx.x * blockDim.x + threadIdx.x;
    if (i < D_MODEL * PATCH) {
        int n = i / PATCH, k = i % PATCH;
        o[i] = (__bf16)w[k * D_MODEL + n];   // conv_w [K,N] -> conv_wT [N,K] bf16
    }
}
__global__ void cvt_f32_bf16(const float* __restrict__ a, __bf16* __restrict__ o, int n) {
    int i = blockIdx.x * blockDim.x + threadIdx.x;
    if (i < n) o[i] = (__bf16)a[i];
}

// ---------------------------------------------------------------------------
// Patchify conv (GEMM M=32000,K=160,N=256) + exact GELU + positional embedding
// Whole 80KB conv weight matrix DMA'd to LDS per WG via TDM; B reads are ds.
// sched_barrier pins: all 10 ds_loads issue first, then 5 chained WMMAs.
// ---------------------------------------------------------------------------
__global__ void __launch_bounds__(256) conv_gelu_pos(
        const float* __restrict__ wav, const __bf16* __restrict__ wT,
        const float* __restrict__ pos, float* __restrict__ x) {
    extern __shared__ __bf16 smw[];   // 256*160 bf16 = 80 KB
    const int tid = threadIdx.x, lane = tid & 31, w = tid >> 5;
    const int col = lane & 15, hh = lane >> 4;
    const int m0 = blockIdx.x * 64;
    const int mtl = w & 3;
    const int ntbase = (w >> 2) * 8;

    if (tid < 32) {   // wave 0 issues the weight DMA (2D tile: 160 x 256 rows)
        tdm_load_bf16(lds_off_u32(smw), (uint64_t)(uintptr_t)wT,
                      PATCH, D_MODEL, 0,            // tile: x=160, y=256 rows
                      PATCH, D_MODEL, 1,            // tensor dims
                      PATCH, 0);                    // row stride 160 elems
        __builtin_amdgcn_s_wait_tensorcnt(0);
    }

    // A fragments: 5 K-tiles of the 16x160 patch block, bf16-converted on load
    v16bf a[5];
    {
        const int m = m0 + mtl * 16 + col;
        const int b = m / SEQ, s = m % SEQ;
        const float* row = wav + (size_t)b * 160000 + (size_t)s * PATCH;
#pragma unroll
        for (int kt = 0; kt < 5; kt++) {
            const float* p0 = row + kt * 32 + hh * 16;
            const float* p1 = p0 + 8;
            v16bf f;
#pragma unroll
            for (int i = 0; i < 8; i++) { f[i] = (__bf16)p0[i]; f[8 + i] = (__bf16)p1[i]; }
            a[kt] = f;
        }
    }
    __syncthreads();   // weights in LDS

    for (int i = 0; i < 8; i++) {
        const int nt = ntbase + i;
        const __bf16* bb = smw + (size_t)(nt * 16 + col) * PATCH + hh * 16;
        v16bf bf[5];
#pragma unroll
        for (int kt = 0; kt < 5; kt++) bf[kt] = *(const v16bf*)(bb + kt * 32);
        __builtin_amdgcn_sched_barrier(0);      // keep loads above, WMMAs below
        v8f acc = {};
#pragma unroll
        for (int kt = 0; kt < 5; kt++)
            acc = __builtin_amdgcn_wmma_f32_16x16x32_bf16(false, a[kt], false, bf[kt],
                                                          (short)0, acc, false, false);
        const int n = nt * 16 + col;
#pragma unroll
        for (int r = 0; r < 8; r++) {
            const int m = m0 + mtl * 16 + hh * 8 + r;
            const int s = m % SEQ;
            x[(size_t)m * D_MODEL + n] = gelu_exact(acc[r]) + pos[s * D_MODEL + n];
        }
    }
}

// ---------------------------------------------------------------------------
// LayerNorm over rows of 256; one wave per row
// ---------------------------------------------------------------------------
__global__ void __launch_bounds__(256) ln_rows_bf16(
        const float* __restrict__ x, __bf16* __restrict__ o,
        const float* __restrict__ scale, const float* __restrict__ bias) {
    const int lane = threadIdx.x & 31;
    const int row = blockIdx.x * 8 + (threadIdx.x >> 5);
    const float* xr = x + (size_t)row * D_MODEL;
    float v[8], s = 0.f, s2 = 0.f;
#pragma unroll
    for (int j = 0; j < 8; j++) { float t = xr[j * 32 + lane]; v[j] = t; s += t; s2 += t * t; }
#pragma unroll
    for (int m = 16; m >= 1; m >>= 1) { s += __shfl_xor(s, m, 32); s2 += __shfl_xor(s2, m, 32); }
    const float mu = s * (1.0f / 256.0f);
    const float rstd = rsqrtf(s2 * (1.0f / 256.0f) - mu * mu + 1e-5f);
    __bf16* orow = o + (size_t)row * D_MODEL;
#pragma unroll
    for (int j = 0; j < 8; j++) {
        int k = j * 32 + lane;
        orow[k] = (__bf16)((v[j] - mu) * rstd * scale[k] + bias[k]);
    }
}

__global__ void __launch_bounds__(256) ln_rows_f32(
        const float* __restrict__ x, float* __restrict__ o,
        const float* __restrict__ scale, const float* __restrict__ bias) {
    const int lane = threadIdx.x & 31;
    const int row = blockIdx.x * 8 + (threadIdx.x >> 5);
    const float* xr = x + (size_t)row * D_MODEL;
    float v[8], s = 0.f, s2 = 0.f;
#pragma unroll
    for (int j = 0; j < 8; j++) { float t = xr[j * 32 + lane]; v[j] = t; s += t; s2 += t * t; }
#pragma unroll
    for (int m = 16; m >= 1; m >>= 1) { s += __shfl_xor(s, m, 32); s2 += __shfl_xor(s2, m, 32); }
    const float mu = s * (1.0f / 256.0f);
    const float rstd = rsqrtf(s2 * (1.0f / 256.0f) - mu * mu + 1e-5f);
    float* orow = o + (size_t)row * D_MODEL;
#pragma unroll
    for (int j = 0; j < 8; j++) {
        int k = j * 32 + lane;
        orow[k] = (v[j] - mu) * rstd * scale[k] + bias[k];
    }
}

// ---------------------------------------------------------------------------
// Input projection GEMM: xp = LN(x) @ w_ih^T + b_ih   (M=32000, K=256, N=768)
// WG covers 128 positions; wave w owns M-tile w (A in VGPRs for whole kernel);
// all waves sweep the 48 N-tiles sharing a double-buffered LDS B tile filled
// with global_load_async_to_lds_b128 (ASYNCcnt pipelined one stage deep).
// sched_barrier pins: 16 ds_loads first, then 8 chained WMMAs.
// ---------------------------------------------------------------------------
__global__ void __launch_bounds__(256) xproj(
        const __bf16* __restrict__ hn, const __bf16* __restrict__ wih,
        const float* __restrict__ bih, float* __restrict__ xp) {
    extern __shared__ __bf16 bsm[];   // 2 buffers * 16*256 bf16 = 16 KB
    const int tid = threadIdx.x, lane = tid & 31, w = tid >> 5;
    const int col = lane & 15, hh = lane >> 4;
    const int m0 = blockIdx.x * 128;

    const uint32_t ldsbase = lds_off_u32(bsm);
    const int rr = tid >> 4, cc = tid & 15;   // 256 threads cover 16 rows x 16 chunks

    // async stage of B tile nt (16 rows x 256 cols bf16 = 8 KB) into buffer buf
    auto stage = [&](int nt, int buf) {
        const __bf16* src = wih + ((size_t)(nt * 16 + rr) << 8) + cc * 16;
        uint32_t dst = ldsbase + (uint32_t)buf * 8192u + (uint32_t)rr * 512u + (uint32_t)cc * 32u;
        asm volatile(
            "global_load_async_to_lds_b128 %0, %1, off\n\t"
            "global_load_async_to_lds_b128 %0, %1, off offset:16"
            :: "v"(dst), "v"(src) : "memory");
    };

    // A fragments: wave w owns M-tile w; held in VGPRs for all 48 N-tiles
    v16bf a[8];
    {
        const __bf16* row = hn + (size_t)(m0 + w * 16 + col) * D_MODEL + hh * 16;
#pragma unroll
        for (int kt = 0; kt < 8; kt++) {
            v8bf lo = *(const v8bf*)(row + kt * 32);
            v8bf hi = *(const v8bf*)(row + kt * 32 + 8);
            a[kt] = __builtin_shufflevector(lo, hi, 0,1,2,3,4,5,6,7,8,9,10,11,12,13,14,15);
        }
    }

    stage(0, 0);
    stage(1, 1);
    for (int i = 0; i < 48; i++) {
        if (i < 47) asm volatile("s_wait_asynccnt 0x2" ::: "memory");
        else        asm volatile("s_wait_asynccnt 0x0" ::: "memory");
        __syncthreads();                        // stage i visible to all waves
        const __bf16* bt = bsm + (size_t)(i & 1) * 4096 + (size_t)col * D_MODEL + hh * 16;
        v16bf bf[8];
#pragma unroll
        for (int kt = 0; kt < 8; kt++) bf[kt] = *(const v16bf*)(bt + kt * 32);
        __builtin_amdgcn_sched_barrier(0);      // keep loads above, WMMAs below
        v8f acc = {};
#pragma unroll
        for (int kt = 0; kt < 8; kt++)
            acc = __builtin_amdgcn_wmma_f32_16x16x32_bf16(false, a[kt], false, bf[kt],
                                                          (short)0, acc, false, false);
        const int n = i * 16 + col;
        const float bi = bih[n];
#pragma unroll
        for (int r = 0; r < 8; r++) {
            const int m = m0 + w * 16 + hh * 8 + r;
            xp[(size_t)m * G3D + n] = acc[r] + bi;
        }
        __syncthreads();                        // everyone done reading buf (i&1)
        if (i + 2 < 48) stage(i + 2, i & 1);
    }
}

// zero the step-0 hidden state buffer and the inter-WG barrier counter
__global__ void scan_init(__bf16* hbuf, int* ctr) {
    int i = blockIdx.x * blockDim.x + threadIdx.x;
    if (i == 0) *ctr = 0;
    if (i < BATCH * D_MODEL) hbuf[i] = (__bf16)0.0f;
}

// ---------------------------------------------------------------------------
// GRU recurrence: 1000 serial steps of h @ w_hh^T. 4 cooperating workgroups,
// each DMA-ing its 3-gate x 64-row x 256-col w_hh slice (96 KB) into LDS via
// one 3D TDM descriptor. h double-buffered in global bf16; atomic grid barrier.
// Per step: 8 A fragments preloaded; B fragments consumed in two half-K
// batches of 12, each pinned load-block -> 12 back-to-back WMMAs.
// ---------------------------------------------------------------------------
__global__ void __launch_bounds__(256) gru_scan(
        const float* __restrict__ xp, float* __restrict__ x,
        const __bf16* __restrict__ whh, const float* __restrict__ bhh,
        __bf16* __restrict__ hbuf, int* __restrict__ ctr) {
    extern __shared__ __bf16 smem[];   // 192*256 bf16 = 96 KB
    const int g = blockIdx.x;          // WG g owns hidden dims [g*64, g*64+64)
    const int tid = threadIdx.x, lane = tid & 31, w = tid >> 5;
    const int col = lane & 15, hh = lane >> 4;

    if (tid < 32) {   // 3D tile: x=256 elems, y=64 rows (stride 256), z=3 gates (stride 256*256)
        tdm_load_bf16(lds_off_u32(smem),
                      (uint64_t)(uintptr_t)(whh + (size_t)g * 64 * D_MODEL),
                      D_MODEL, 64, 3,
                      D_MODEL, G3D, 3,
                      D_MODEL, (uint64_t)D_MODEL * D_MODEL);
        __builtin_amdgcn_s_wait_tensorcnt(0);
    }
    __syncthreads();

    const int mt = w >> 2, ntd = w & 3;
    const int d = g * 64 + ntd * 16 + col;          // hidden dim (C columns)
    const float bhr = bhh[d], bhz = bhh[256 + d], bhn = bhh[512 + d];
    const int ma = mt * 16 + col;                   // A-fragment row (batch)
    const __bf16* lbbase = smem + (size_t)(ntd * 16 + col) * D_MODEL + hh * 16;
    v8f hprev = {};

    for (int s = 0; s < SEQ; s++) {
        // preload all 8 A fragments for this step
        const __bf16* abase = hbuf + (size_t)(s & 1) * (BATCH * D_MODEL)
                                   + (size_t)ma * D_MODEL + hh * 16;
        v16bf a[8];
#pragma unroll
        for (int kt = 0; kt < 8; kt++) {
            v8bf lo = *(const v8bf*)(abase + kt * 32);
            v8bf hi = *(const v8bf*)(abase + kt * 32 + 8);
            a[kt] = __builtin_shufflevector(lo, hi, 0,1,2,3,4,5,6,7,8,9,10,11,12,13,14,15);
        }
        v8f cr = {}, cz = {}, cn = {};
#pragma unroll
        for (int half = 0; half < 2; half++) {
            v16bf br[4], bz[4], bn[4];
#pragma unroll
            for (int q = 0; q < 4; q++) {
                const __bf16* lb = lbbase + (half * 4 + q) * 32;
                br[q] = *(const v16bf*)(lb);
                bz[q] = *(const v16bf*)(lb + 64 * D_MODEL);
                bn[q] = *(const v16bf*)(lb + 128 * D_MODEL);
            }
            __builtin_amdgcn_sched_barrier(0);  // loads above, WMMA chain below
#pragma unroll
            for (int q = 0; q < 4; q++) {
                const int kt = half * 4 + q;
                cr = __builtin_amdgcn_wmma_f32_16x16x32_bf16(false, a[kt], false, br[q], (short)0, cr, false, false);
                cz = __builtin_amdgcn_wmma_f32_16x16x32_bf16(false, a[kt], false, bz[q], (short)0, cz, false, false);
                cn = __builtin_amdgcn_wmma_f32_16x16x32_bf16(false, a[kt], false, bn[q], (short)0, cn, false, false);
            }
        }
        __bf16* hout = hbuf + (size_t)((s + 1) & 1) * (BATCH * D_MODEL);
#pragma unroll
        for (int r = 0; r < 8; r++) {
            const int m = mt * 16 + hh * 8 + r;            // batch index
            const size_t xb = ((size_t)m * SEQ + s) * G3D;
            const float rg = sigmoid_f(xp[xb + d]        + cr[r] + bhr);
            const float zg = sigmoid_f(xp[xb + 256 + d]  + cz[r] + bhz);
            const float ng = tanhf(    xp[xb + 512 + d]  + rg * (cn[r] + bhn));
            const float hv = (1.0f - zg) * ng + zg * hprev[r];
            hprev[r] = hv;
            hout[(size_t)m * D_MODEL + d] = (__bf16)hv;
            const size_t xi = ((size_t)m * SEQ + s) * D_MODEL + d;
            x[xi] = x[xi] + hv;                            // residual, in place
        }
        // ---- 4-WG grid barrier (monotonic counter; grid=4 => co-resident)
        __threadfence();
        __syncthreads();
        if (tid == 0) {
            atomicAdd(ctr, 1);
            const int target = 4 * (s + 1);
            while (__hip_atomic_load(ctr, __ATOMIC_ACQUIRE, __HIP_MEMORY_SCOPE_AGENT) < target)
                __builtin_amdgcn_s_sleep(1);
        }
        __syncthreads();
        __threadfence();
    }
}

// ---------------------------------------------------------------------------
__global__ void __launch_bounds__(256) mean_rows(const float* __restrict__ t,
                                                 float* __restrict__ emb) {
    const int b = blockIdx.x, dcol = threadIdx.x;
    float acc = 0.0f;
    const float* p = t + (size_t)b * SEQ * D_MODEL + dcol;
    for (int s = 0; s < SEQ; s++) acc += p[(size_t)s * D_MODEL];
    emb[b * D_MODEL + dcol] = acc * (1.0f / SEQ);
}

__global__ void __launch_bounds__(256) head_mlp(
        const float* __restrict__ emb, const float* __restrict__ w1,
        const float* __restrict__ b1, const float* __restrict__ w2,
        const float* __restrict__ b2, float* __restrict__ out) {
    __shared__ float se[BATCH * D_MODEL];
    __shared__ float sh[BATCH * 128];
    const int tid = threadIdx.x;
    for (int i = tid; i < BATCH * D_MODEL; i += 256) se[i] = emb[i];
    __syncthreads();
    for (int i = tid; i < BATCH * 128; i += 256) {
        const int b = i >> 7, j = i & 127;
        float acc = b1[j];
        for (int k = 0; k < D_MODEL; k++) acc += se[b * D_MODEL + k] * w1[k * 128 + j];
        sh[i] = gelu_exact(acc);
    }
    __syncthreads();
    if (tid < BATCH * 8) {
        const int b = tid >> 3, c = tid & 7;
        float acc = b2[c];
        for (int j = 0; j < 128; j++) acc += sh[b * 128 + j] * w2[j * 8 + c];
        out[tid] = acc;
    }
}

// ---------------------------------------------------------------------------
extern "C" void kernel_launch(void* const* d_in, const int* in_sizes, int n_in,
                              void* d_out, int out_size, void* d_ws, size_t ws_size,
                              hipStream_t stream) {
    const float* wav   = (const float*)d_in[0];
    const float* convw = (const float*)d_in[1];
    const float* pos   = (const float*)d_in[2];
    const float* lns   = (const float*)d_in[3];
    const float* lnb   = (const float*)d_in[4];
    const float* wih   = (const float*)d_in[5];
    const float* whh   = (const float*)d_in[6];
    const float* bih   = (const float*)d_in[7];
    const float* bhh   = (const float*)d_in[8];
    const float* fns   = (const float*)d_in[9];
    const float* fnb   = (const float*)d_in[10];
    const float* hw1   = (const float*)d_in[11];
    const float* hb1   = (const float*)d_in[12];
    const float* hw2   = (const float*)d_in[13];
    const float* hb2   = (const float*)d_in[14];
    float* out = (float*)d_out;
    (void)in_sizes; (void)n_in; (void)out_size; (void)ws_size;

    char* ws = (char*)d_ws;
    size_t off = 0;
    auto alloc = [&](size_t bytes) -> void* {
        void* p = ws + off;
        off = (off + bytes + 255) & ~(size_t)255;
        return p;
    };
    float*  x     = (float*) alloc((size_t)NPOS * D_MODEL * 4);
    __bf16* hnorm = (__bf16*)alloc((size_t)NPOS * D_MODEL * 2);
    float*  xp    = (float*) alloc((size_t)NPOS * G3D * 4);
    __bf16* cwT   = (__bf16*)alloc((size_t)D_MODEL * PATCH * 2);
    __bf16* wihb  = (__bf16*)alloc((size_t)NLAYERS * G3D * D_MODEL * 2);
    __bf16* whhb  = (__bf16*)alloc((size_t)NLAYERS * G3D * D_MODEL * 2);
    __bf16* hbuf  = (__bf16*)alloc((size_t)2 * BATCH * D_MODEL * 2);
    float*  emb   = (float*) alloc((size_t)BATCH * D_MODEL * 4);
    int*    ctr   = (int*)   alloc(256);

    cvt_convw<<<(D_MODEL * PATCH + 255) / 256, 256, 0, stream>>>(convw, cwT);
    const int gateN = NLAYERS * G3D * D_MODEL;
    cvt_f32_bf16<<<(gateN + 255) / 256, 256, 0, stream>>>(wih, wihb, gateN);
    cvt_f32_bf16<<<(gateN + 255) / 256, 256, 0, stream>>>(whh, whhb, gateN);

    conv_gelu_pos<<<NPOS / 64, 256, D_MODEL * PATCH * sizeof(__bf16), stream>>>(
        wav, cwT, pos, x);

    for (int l = 0; l < NLAYERS; l++) {
        ln_rows_bf16<<<NPOS / 8, 256, 0, stream>>>(x, hnorm,
                                                   lns + l * D_MODEL, lnb + l * D_MODEL);
        xproj<<<NPOS / 128, 256, 2 * 16 * D_MODEL * sizeof(__bf16), stream>>>(
            hnorm, wihb + (size_t)l * G3D * D_MODEL, bih + l * G3D, xp);
        scan_init<<<(BATCH * D_MODEL + 255) / 256, 256, 0, stream>>>(hbuf, ctr);
        gru_scan<<<4, 256, 192 * D_MODEL * sizeof(__bf16), stream>>>(
            xp, x, whhb + (size_t)l * G3D * D_MODEL, bhh + l * G3D, hbuf, ctr);
    }

    float* lnf = xp;  // reuse xp region for final-LN rows
    ln_rows_f32<<<NPOS / 8, 256, 0, stream>>>(x, lnf, fns, fnb);
    mean_rows<<<BATCH, 256, 0, stream>>>(lnf, emb);
    head_mlp<<<1, 256, 0, stream>>>(emb, hw1, hb1, hw2, hb2, out);
}
